// LanguageMamba_24859270709414
// MI455X (gfx1250) — compile-verified, matrix operand order checked by reading, not run on hardware
//
#include <hip/hip_runtime.h>

// ---------------------------------------------------------------------------
// Model constants (from the reference)
// ---------------------------------------------------------------------------
constexpr int V_   = 50257;
constexpr int D_   = 768;
constexpr int NL_  = 12;
constexpr int DI_  = 1536;
constexpr int DS_  = 16;     // state dim
constexpr int DC_  = 4;      // conv width
constexpr int DTR_ = 48;     // dt rank
constexpr int B_   = 2;
constexpr int S_   = 1024;
constexpr int M_   = B_ * S_;        // 2048 activation rows
constexpr int XZ_  = 2 * DI_;        // 3072
constexpr int DBL_ = DTR_ + 2 * DS_; // 80
constexpr int KP_  = 64;             // dt rank padded to WMMA K multiple
constexpr int NC_  = 16;             // scan chunks
constexpr int CL_  = S_ / NC_;       // 64 steps per chunk

// GEMM tiling
constexpr int BM_   = 128;           // block rows
constexpr int BN_   = 128;           // block cols
constexpr int BK_   = 64;            // K slab
constexpr int PADW  = 72;            // padded LDS row (bf16 elems) -> 144B, bank-conflict free
constexpr unsigned TILE_B   = (unsigned)(BM_ * PADW * 2);   // bytes per A (or B) tile: 18432
constexpr unsigned BUF_B    = 2u * TILE_B;                  // A+B per buffer: 36864

typedef __attribute__((ext_vector_type(16))) __bf16 v16bf;
typedef __attribute__((ext_vector_type(8)))  __bf16 v8bf;
typedef __attribute__((ext_vector_type(8)))  float  v8f;

__device__ __forceinline__ __bf16 f2bf(float f) {
  unsigned u = __builtin_bit_cast(unsigned, f);
  u += 0x7FFFu + ((u >> 16) & 1u);              // round-to-nearest-even
  unsigned short h = (unsigned short)(u >> 16);
  return __builtin_bit_cast(__bf16, h);
}

__device__ __forceinline__ v8f v8f_zero() {
  v8f z = {0.f, 0.f, 0.f, 0.f, 0.f, 0.f, 0.f, 0.f};
  return z;
}

// ---------------------------------------------------------------------------
// Async copy of a BM_ x BK_ bf16 tile (row-major, leading dim ld) into LDS at
// byte offset lds_base, padded row stride PADW. 256 threads, 4 x 16B chunks
// each, via gfx1250 GLOBAL_LOAD_ASYNC_TO_LDS_B128 (tracked by ASYNCcnt).
// ---------------------------------------------------------------------------
__device__ __forceinline__ void async_tile(const __bf16* __restrict__ g, int row0,
                                           int ld, int k, int rowmax,
                                           unsigned lds_base, int tid) {
#pragma unroll
  for (int i = 0; i < 4; ++i) {
    int id = tid + i * 256;          // 0..1023
    int r  = id >> 3;                // tile row 0..127
    int cc = id & 7;                 // 16B chunk within the 64-elem row
    int gr = row0 + r; gr = (gr > rowmax) ? rowmax : gr;   // clamp ragged N edge
    const __bf16* gp = g + (long)gr * ld + k + cc * 8;
    unsigned loff = lds_base + (unsigned)(r * (PADW * 2) + cc * 16);
    asm volatile("global_load_async_to_lds_b128 %0, %1, off"
                 :: "v"(loff), "v"(gp) : "memory");
  }
}

// WMMA 16x32 bf16 fragment from an LDS tile (padded row stride PADW).
// CDNA5 16-bit A/B layout: lanes 0-15 rows, K 0-7/16-23; lanes 16-31 K 8-15/24-31.
__device__ __forceinline__ v16bf lds_frag(const __bf16* __restrict__ sm, int row0, int ks) {
  const int lane = threadIdx.x & 31;
  const int r  = row0 + (lane & 15);
  const int kh = lane >> 4;
  const __bf16* p = sm + r * PADW + ks + kh * 8;
  v8bf lo = *(const v8bf*)(p);
  v8bf hi = *(const v8bf*)(p + 16);
  v16bf f;
#pragma unroll
  for (int i = 0; i < 8; ++i) { f[i] = lo[i]; f[8 + i] = hi[i]; }
  return f;
}

// ---------------------------------------------------------------------------
// GEMM: C[M,N] (f32, optional bf16 mirror) = A[M,K] @ B[N,K]^T, bf16 inputs.
// 128x128 block, BK=64 double-buffered in LDS via async-to-LDS copies.
// 8 waves; each wave owns a 32x64 tile = 2x4 WMMA accumulators.
// Requires: M % 128 == 0, K % 64 == 0, lda/ldb % 8 == 0 (all hold here).
// ---------------------------------------------------------------------------
__global__ __launch_bounds__(256) void k_gemm(const __bf16* __restrict__ A,
                                              const __bf16* __restrict__ Bw,
                                              float* __restrict__ C,
                                              __bf16* __restrict__ Cbf,
                                              int M, int N, int K,
                                              int lda, int ldb, int ldc) {
  __shared__ __bf16 smem[2 * 2 * BM_ * PADW];   // [buf][A|B][128][72] = 72KB

  const int tid  = threadIdx.x;
  const int lane = tid & 31;
  const int wid  = tid >> 5;
  const int wm   = (wid & 3) * 32;              // wave row offset in block
  const int wn   = (wid >> 2) * 64;             // wave col offset in block
  const int mB   = blockIdx.y * BM_;
  const int nB   = blockIdx.x * BN_;

  const unsigned sbase = (unsigned)(unsigned long long)(void*)smem;  // LDS byte offset

  v8f c[2][4];
#pragma unroll
  for (int i = 0; i < 2; ++i)
#pragma unroll
    for (int j = 0; j < 4; ++j) c[i][j] = v8f_zero();

  // prologue: fill buffer 0
  async_tile(A,  mB, lda, 0, M - 1, sbase,          tid);
  async_tile(Bw, nB, ldb, 0, N - 1, sbase + TILE_B, tid);
  asm volatile("s_wait_asynccnt 0" ::: "memory");
  __syncthreads();

  int buf = 0;
  for (int kt = 0; kt < K; kt += BK_) {
    if (kt + BK_ < K) {                         // stage next slab into other buffer
      unsigned nb = sbase + (unsigned)(buf ^ 1) * BUF_B;
      async_tile(A,  mB, lda, kt + BK_, M - 1, nb,          tid);
      async_tile(Bw, nB, ldb, kt + BK_, N - 1, nb + TILE_B, tid);
    }

    const __bf16* As = smem + (size_t)buf * (2 * BM_ * PADW);
    const __bf16* Bs = As + BM_ * PADW;

#pragma unroll
    for (int ks = 0; ks < BK_; ks += 32) {
      v16bf a0 = lds_frag(As, wm,      ks);
      v16bf a1 = lds_frag(As, wm + 16, ks);
      v16bf b0 = lds_frag(Bs, wn +  0, ks);
      v16bf b1 = lds_frag(Bs, wn + 16, ks);
      v16bf b2 = lds_frag(Bs, wn + 32, ks);
      v16bf b3 = lds_frag(Bs, wn + 48, ks);
      c[0][0] = __builtin_amdgcn_wmma_f32_16x16x32_bf16(false, a0, false, b0, (short)0, c[0][0], false, false);
      c[0][1] = __builtin_amdgcn_wmma_f32_16x16x32_bf16(false, a0, false, b1, (short)0, c[0][1], false, false);
      c[0][2] = __builtin_amdgcn_wmma_f32_16x16x32_bf16(false, a0, false, b2, (short)0, c[0][2], false, false);
      c[0][3] = __builtin_amdgcn_wmma_f32_16x16x32_bf16(false, a0, false, b3, (short)0, c[0][3], false, false);
      c[1][0] = __builtin_amdgcn_wmma_f32_16x16x32_bf16(false, a1, false, b0, (short)0, c[1][0], false, false);
      c[1][1] = __builtin_amdgcn_wmma_f32_16x16x32_bf16(false, a1, false, b1, (short)0, c[1][1], false, false);
      c[1][2] = __builtin_amdgcn_wmma_f32_16x16x32_bf16(false, a1, false, b2, (short)0, c[1][2], false, false);
      c[1][3] = __builtin_amdgcn_wmma_f32_16x16x32_bf16(false, a1, false, b3, (short)0, c[1][3], false, false);
    }

    asm volatile("s_wait_asynccnt 0" ::: "memory");  // own async copies landed
    __syncthreads();                                 // whole block's copies + reads done
    buf ^= 1;
  }

  // C/D layout: lanes 0-15 col=lane rows 0..7 per VGPR; lanes 16-31 rows 8..15.
#pragma unroll
  for (int i = 0; i < 2; ++i) {
    const int rowb = mB + wm + 16 * i + (lane >> 4) * 8;
#pragma unroll
    for (int j = 0; j < 4; ++j) {
      const int col = nB + wn + 16 * j + (lane & 15);
      if (col < N) {
#pragma unroll
        for (int r = 0; r < 8; ++r) {
          const long off = (long)(rowb + r) * ldc + col;
          C[off] = c[i][j][r];
          if (Cbf) Cbf[off] = f2bf(c[i][j][r]);
        }
      }
    }
  }
}

// ---------------------------------------------------------------------------
// Elementwise / support kernels
// ---------------------------------------------------------------------------
__global__ void k_cvt(const float* __restrict__ in, __bf16* __restrict__ out, long n) {
  long i = (long)blockIdx.x * blockDim.x + threadIdx.x;
  if (i < n) out[i] = f2bf(in[i]);
}

__global__ void k_cvt_pad(const float* __restrict__ in, __bf16* __restrict__ out,
                          int rows, int in_stride, int kcopy, int kout) {
  long i = (long)blockIdx.x * blockDim.x + threadIdx.x;
  if (i >= (long)rows * kout) return;
  int r = (int)(i / kout), k = (int)(i % kout);
  float v = (k < kcopy) ? in[(long)r * in_stride + k] : 0.f;
  out[i] = f2bf(v);
}

__global__ void k_embed(const int* __restrict__ ids, const float* __restrict__ emb,
                        float* __restrict__ x, __bf16* __restrict__ xbf) {
  long i = (long)blockIdx.x * blockDim.x + threadIdx.x;
  if (i >= (long)M_ * D_) return;
  int row = (int)(i / D_), col = (int)(i % D_);
  float v = emb[(long)ids[row] * D_ + col] * 27.712812921102035f;  // sqrt(768)
  x[i] = v;
  xbf[i] = f2bf(v);
}

__global__ void k_conv(const float* __restrict__ xz, const float* __restrict__ cw,
                       const float* __restrict__ cb, float* __restrict__ xin,
                       __bf16* __restrict__ xinbf) {
  long i = (long)blockIdx.x * blockDim.x + threadIdx.x;
  if (i >= (long)M_ * DI_) return;
  int d  = (int)(i % DI_);
  int ts = (int)(i / DI_);        // b*S + t
  int t  = ts % S_;
  float acc = cb[d];
#pragma unroll
  for (int j = 0; j < DC_; ++j) {
    int tt = t - (DC_ - 1) + j;
    if (tt >= 0) acc += cw[d * DC_ + j] * xz[(long)(ts - (DC_ - 1) + j) * XZ_ + d];
  }
  float s = acc * (1.f / (1.f + __expf(-acc)));  // SiLU
  xin[i] = s;
  xinbf[i] = f2bf(s);
}

__global__ void k_softplus_bias(float* __restrict__ v, const float* __restrict__ db) {
  long i = (long)blockIdx.x * blockDim.x + threadIdx.x;
  if (i >= (long)M_ * DI_) return;
  float x = v[i] + db[(int)(i % DI_)];
  v[i] = (x > 20.f) ? x : log1pf(__expf(x));
}

// ---------------------------------------------------------------------------
// Chunked selective scan (linear diagonal recurrence => associative).
// Pass 1: per (b,d,chunk) compute chunk transfer: P = prod exp(dt*A), q = state
//         reached from h=0. Pass 2: per (b,d) compose 16 chunk states. Pass 3:
//         replay each chunk from its entry state, fusing D-skip + SiLU(z) gate.
// ---------------------------------------------------------------------------
__global__ void k_scan_p1(const float* __restrict__ dt, const float* __restrict__ xin,
                          const float* __restrict__ dbl, const float* __restrict__ Alog,
                          float* __restrict__ Pv, float* __restrict__ Qv) {
  int idx = blockIdx.x * blockDim.x + threadIdx.x;
  if (idx >= B_ * DI_ * NC_) return;
  const int c = idx % NC_;
  const int d = (idx / NC_) % DI_;
  const int b = idx / (NC_ * DI_);

  float Arow[DS_];
#pragma unroll
  for (int s = 0; s < DS_; ++s) Arow[s] = -__expf(Alog[(long)d * DS_ + s]);

  float P[DS_], h[DS_];
#pragma unroll
  for (int s = 0; s < DS_; ++s) { P[s] = 1.f; h[s] = 0.f; }

  const int t0 = c * CL_;
  for (int t = 0; t < CL_; ++t) {
    const long base = (long)b * S_ + t0 + t;
    const float dtv = dt[base * DI_ + d];
    const float dx  = dtv * xin[base * DI_ + d];
    const float* bc = dbl + base * DBL_;
#pragma unroll
    for (int s = 0; s < DS_; ++s) {
      float e = __expf(dtv * Arow[s]);
      P[s] *= e;
      h[s] = h[s] * e + dx * bc[DTR_ + s];
    }
  }
  const long o = ((long)(b * DI_ + d) * NC_ + c) * DS_;
#pragma unroll
  for (int s = 0; s < DS_; ++s) { Pv[o + s] = P[s]; Qv[o + s] = h[s]; }
}

__global__ void k_scan_p2(const float* __restrict__ Pv, const float* __restrict__ Qv,
                          float* __restrict__ Hin) {
  int idx = blockIdx.x * blockDim.x + threadIdx.x;
  if (idx >= B_ * DI_) return;
  const long base = (long)idx * NC_ * DS_;
  float h[DS_];
#pragma unroll
  for (int s = 0; s < DS_; ++s) h[s] = 0.f;
  for (int c = 0; c < NC_; ++c) {
    const long o = base + (long)c * DS_;
#pragma unroll
    for (int s = 0; s < DS_; ++s) {
      Hin[o + s] = h[s];                       // state entering chunk c
      h[s] = Pv[o + s] * h[s] + Qv[o + s];
    }
  }
}

__global__ void k_scan_p3(const float* __restrict__ dt, const float* __restrict__ xin,
                          const float* __restrict__ dbl, const float* __restrict__ xz,
                          const float* __restrict__ Alog, const float* __restrict__ Dp,
                          const float* __restrict__ Hin, __bf16* __restrict__ ybf) {
  int idx = blockIdx.x * blockDim.x + threadIdx.x;
  if (idx >= B_ * DI_ * NC_) return;
  const int c = idx % NC_;
  const int d = (idx / NC_) % DI_;
  const int b = idx / (NC_ * DI_);

  float Arow[DS_];
#pragma unroll
  for (int s = 0; s < DS_; ++s) Arow[s] = -__expf(Alog[(long)d * DS_ + s]);
  const float Dd = Dp[d];

  float h[DS_];
  const long o = ((long)(b * DI_ + d) * NC_ + c) * DS_;
#pragma unroll
  for (int s = 0; s < DS_; ++s) h[s] = Hin[o + s];

  const int t0 = c * CL_;
  for (int t = 0; t < CL_; ++t) {
    const long base = (long)b * S_ + t0 + t;
    const float dtv = dt[base * DI_ + d];
    const float xv  = xin[base * DI_ + d];
    const float zv  = xz[base * XZ_ + DI_ + d];
    const float* bc = dbl + base * DBL_;
    const float dx  = dtv * xv;
    float y = 0.f;
#pragma unroll
    for (int s = 0; s < DS_; ++s) {
      float hn = h[s] * __expf(dtv * Arow[s]) + dx * bc[DTR_ + s];
      h[s] = hn;
      y += hn * bc[DTR_ + DS_ + s];
    }
    y += xv * Dd;
    float sig = 1.f / (1.f + __expf(-zv));
    ybf[base * DI_ + d] = f2bf(y * zv * sig);
  }
}

// LayerNorm: one wave32 per row, shuffle reduction.
__global__ void k_layernorm(const float* __restrict__ x, const float* __restrict__ w,
                            const float* __restrict__ b, __bf16* __restrict__ out) {
  const int lane = threadIdx.x & 31;
  const int row  = blockIdx.x * (blockDim.x >> 5) + (threadIdx.x >> 5);
  if (row >= M_) return;
  const float* xr = x + (long)row * D_;
  float s = 0.f, s2 = 0.f;
  for (int j = lane; j < D_; j += 32) { float v = xr[j]; s += v; s2 += v * v; }
#pragma unroll
  for (int o = 16; o > 0; o >>= 1) { s += __shfl_xor(s, o, 32); s2 += __shfl_xor(s2, o, 32); }
  const float mu  = s / D_;
  const float inv = rsqrtf(s2 / D_ - mu * mu + 1e-5f);
  for (int j = lane; j < D_; j += 32)
    out[(long)row * D_ + j] = f2bf((xr[j] - mu) * inv * w[j] + b[j]);
}

// ---------------------------------------------------------------------------
// Host launch
// ---------------------------------------------------------------------------
extern "C" void kernel_launch(void* const* d_in, const int* in_sizes, int n_in,
                              void* d_out, int out_size, void* d_ws, size_t ws_size,
                              hipStream_t stream) {
  (void)in_sizes; (void)n_in; (void)out_size; (void)ws_size;

  const int*   ids    = (const int*)  d_in[0];
  const float* emb    = (const float*)d_in[1];
  const float* inw    = (const float*)d_in[2];   // (NL, 3072, 768)
  const float* convw  = (const float*)d_in[3];   // (NL, 1536, 4)
  const float* convb  = (const float*)d_in[4];   // (NL, 1536)
  const float* xprojw = (const float*)d_in[5];   // (NL, 80, 1536)
  const float* dtw    = (const float*)d_in[6];   // (NL, 1536, 48)
  const float* dtbias = (const float*)d_in[7];   // (NL, 1536)
  const float* Alog   = (const float*)d_in[8];   // (NL, 1536, 16)
  const float* Dpar   = (const float*)d_in[9];   // (NL, 1536)
  const float* outw   = (const float*)d_in[10];  // (NL, 768, 1536)
  const float* nw     = (const float*)d_in[11];
  const float* nb     = (const float*)d_in[12];
  float* out = (float*)d_out;

  char* ws = (char*)d_ws;
  size_t off = 0;
  auto alloc = [&](size_t bytes) -> char* {
    char* p = ws + off;
    off = (off + bytes + 255) & ~(size_t)255;
    return p;
  };

  // bf16 weight mirrors (rebuilt every launch: deterministic, no cached state)
  __bf16* emb_bf = (__bf16*)alloc((size_t)V_ * D_ * 2);
  __bf16* inw_bf = (__bf16*)alloc((size_t)NL_ * XZ_ * D_ * 2);
  __bf16* xpw_bf = (__bf16*)alloc((size_t)NL_ * DBL_ * DI_ * 2);
  __bf16* dtw_bf = (__bf16*)alloc((size_t)NL_ * DI_ * KP_ * 2);
  __bf16* opw_bf = (__bf16*)alloc((size_t)NL_ * D_ * DI_ * 2);
  // activations
  float*  x_f    = (float*) alloc((size_t)M_ * D_ * 4);
  __bf16* x_bf   = (__bf16*)alloc((size_t)M_ * D_ * 2);
  float*  xz     = (float*) alloc((size_t)M_ * XZ_ * 4);
  float*  xin    = (float*) alloc((size_t)M_ * DI_ * 4);
  __bf16* xin_bf = (__bf16*)alloc((size_t)M_ * DI_ * 2);
  float*  dbl    = (float*) alloc((size_t)M_ * DBL_ * 4);
  __bf16* dtb_bf = (__bf16*)alloc((size_t)M_ * KP_ * 2);
  float*  dtv    = (float*) alloc((size_t)M_ * DI_ * 4);
  __bf16* y_bf   = (__bf16*)alloc((size_t)M_ * DI_ * 2);
  __bf16* xn_bf  = (__bf16*)alloc((size_t)M_ * D_ * 2);
  // scan chunk state
  float*  Pv     = (float*) alloc((size_t)B_ * DI_ * NC_ * DS_ * 4);
  float*  Qv     = (float*) alloc((size_t)B_ * DI_ * NC_ * DS_ * 4);
  float*  Hin    = (float*) alloc((size_t)B_ * DI_ * NC_ * DS_ * 4);

  auto cdiv = [](long a, long b) { return (int)((a + b - 1) / b); };
  const dim3 blk(256);

  // Weight conversions
  k_cvt<<<cdiv((long)V_ * D_, 256), blk, 0, stream>>>(emb, emb_bf, (long)V_ * D_);
  k_cvt<<<cdiv((long)NL_ * XZ_ * D_, 256), blk, 0, stream>>>(inw, inw_bf, (long)NL_ * XZ_ * D_);
  k_cvt<<<cdiv((long)NL_ * DBL_ * DI_, 256), blk, 0, stream>>>(xprojw, xpw_bf, (long)NL_ * DBL_ * DI_);
  k_cvt<<<cdiv((long)NL_ * D_ * DI_, 256), blk, 0, stream>>>(outw, opw_bf, (long)NL_ * D_ * DI_);
  k_cvt_pad<<<cdiv((long)NL_ * DI_ * KP_, 256), blk, 0, stream>>>(dtw, dtw_bf, NL_ * DI_, DTR_, DTR_, KP_);

  // Embedding gather + scale
  k_embed<<<cdiv((long)M_ * D_, 256), blk, 0, stream>>>(ids, emb, x_f, x_bf);

  for (int i = 0; i < NL_; ++i) {
    const __bf16* wi  = inw_bf + (size_t)i * XZ_ * D_;
    const __bf16* wxp = xpw_bf + (size_t)i * DBL_ * DI_;
    const __bf16* wdt = dtw_bf + (size_t)i * DI_ * KP_;
    const __bf16* wop = opw_bf + (size_t)i * D_ * DI_;

    // xz = x @ in_proj^T   (2048 x 3072, K=768)
    k_gemm<<<dim3(cdiv(XZ_, BN_), M_ / BM_), blk, 0, stream>>>(
        x_bf, wi, xz, nullptr, M_, XZ_, D_, D_, D_, XZ_);

    // depthwise conv + bias + SiLU
    k_conv<<<cdiv((long)M_ * DI_, 256), blk, 0, stream>>>(
        xz, convw + (size_t)i * DI_ * DC_, convb + (size_t)i * DI_, xin, xin_bf);

    // dbl = xin @ x_proj^T   (2048 x 80, K=1536)
    k_gemm<<<dim3(cdiv(DBL_, BN_), M_ / BM_), blk, 0, stream>>>(
        xin_bf, wxp, dbl, nullptr, M_, DBL_, DI_, DI_, DI_, DBL_);

    // dt input (first 48 cols of dbl) -> bf16, K padded to 64
    k_cvt_pad<<<cdiv((long)M_ * KP_, 256), blk, 0, stream>>>(dbl, dtb_bf, M_, DBL_, DTR_, KP_);

    // dt_raw = dtb @ dt_proj^T   (2048 x 1536, K=64)
    k_gemm<<<dim3(cdiv(DI_, BN_), M_ / BM_), blk, 0, stream>>>(
        dtb_bf, wdt, dtv, nullptr, M_, DI_, KP_, KP_, KP_, DI_);

    // dt = softplus(dt_raw + bias)
    k_softplus_bias<<<cdiv((long)M_ * DI_, 256), blk, 0, stream>>>(dtv, dtbias + (size_t)i * DI_);

    // chunked selective scan
    k_scan_p1<<<cdiv((long)B_ * DI_ * NC_, 256), blk, 0, stream>>>(
        dtv, xin, dbl, Alog + (size_t)i * DI_ * DS_, Pv, Qv);
    k_scan_p2<<<cdiv((long)B_ * DI_, 256), blk, 0, stream>>>(Pv, Qv, Hin);
    k_scan_p3<<<cdiv((long)B_ * DI_ * NC_, 256), blk, 0, stream>>>(
        dtv, xin, dbl, xz, Alog + (size_t)i * DI_ * DS_, Dpar + (size_t)i * DI_, Hin, y_bf);

    // x = y @ out_proj^T   (2048 x 768, K=1536); write f32 + bf16 for next layer
    k_gemm<<<dim3(cdiv(D_, BN_), M_ / BM_), blk, 0, stream>>>(
        y_bf, wop, x_f, x_bf, M_, D_, DI_, DI_, DI_, D_);
  }

  // LayerNorm -> bf16
  k_layernorm<<<M_ / 8, blk, 0, stream>>>(x_f, nw, nb, xn_bf);

  // logits = xn @ emb^T   (2048 x 50257, K=768)
  k_gemm<<<dim3(cdiv(V_, BN_), M_ / BM_), blk, 0, stream>>>(
      xn_bf, emb_bf, out, nullptr, M_, V_, D_, D_, D_, V_);
}
